// LSTMold_24129126269610
// MI455X (gfx1250) — matrix operand order repaired and checked by reading, not compiled, not run on hardware
//
#include <hip/hip_runtime.h>
#include <stdint.h>

// ---------------- types ----------------
typedef __bf16 bf16_t;
typedef __bf16  v16bf __attribute__((ext_vector_type(16)));
typedef __bf16  v8bf  __attribute__((ext_vector_type(8)));
typedef float   v8f   __attribute__((ext_vector_type(8)));
typedef unsigned int u32x4 __attribute__((ext_vector_type(4)));
typedef int     i32x8 __attribute__((ext_vector_type(8)));
typedef int     i32x4 __attribute__((ext_vector_type(4)));

#define T_STEPS 512
#define BATCH   64
#define D_INP   512
#define D_LAT   1024
#define KTOT    1536          // D_INP + D_LAT
#define NWG     64            // persistent workgroups (one per WGP)
#define SLICE   16            // latent columns per WG (NWG*SLICE == D_LAT)
#define NTHREADS 128          // 4 waves of 32

__device__ __forceinline__ float fast_sigmoid(float x) {
  return 1.0f / (1.0f + __expf(-x));
}
__device__ __forceinline__ float fast_tanh(float x) {
  return 1.0f - 2.0f / (__expf(2.0f * x) + 1.0f);
}

// A-tile fragment: lane holds K chunk [p, p+16)
__device__ __forceinline__ v16bf load_a16(const bf16_t* p) {
  return *(const v16bf*)p;
}
__device__ __forceinline__ v16bf load_b16(const bf16_t* p) {
  return *(const v16bf*)p;
}
// RA = reuse-A hint: legal when previous wmma used the identical A operand
template <bool RA>
__device__ __forceinline__ v8f wmma_bf16(v16bf a, v16bf b, v8f c) {
  return __builtin_amdgcn_wmma_f32_16x16x32_bf16(false, a, false, b, (short)0, c, RA, false);
}

// ---------------- prologue: f32 -> bf16 conversion of x ----------------
__global__ void lstm455_cvt_x(const float* __restrict__ x, bf16_t* __restrict__ xb, int n4) {
  int stride = gridDim.x * blockDim.x;
  for (int i = blockIdx.x * blockDim.x + threadIdx.x; i < n4; i += stride) {
    float4 v = ((const float4*)x)[i];
    bf16_t* d = xb + (size_t)i * 4;
    d[0] = (bf16_t)v.x; d[1] = (bf16_t)v.y; d[2] = (bf16_t)v.z; d[3] = (bf16_t)v.w;
  }
}

// ---------------- prologue: pack Wf/Wi/Wc/Wo -> bf16 [4][1024][1536], biases -> f32[4096] ----
__global__ void lstm455_pack_w(const float* __restrict__ Wf, const float* __restrict__ Wi,
                               const float* __restrict__ Wc, const float* __restrict__ Wo,
                               const float* __restrict__ bf_, const float* __restrict__ bi_,
                               const float* __restrict__ bc_, const float* __restrict__ bo_,
                               bf16_t* __restrict__ Wout, float* __restrict__ bout) {
  const size_t per = (size_t)D_LAT * KTOT;
  size_t i0 = (size_t)blockIdx.x * blockDim.x + threadIdx.x;
  size_t stride = (size_t)gridDim.x * blockDim.x;
  for (size_t idx = i0; idx < 4 * per; idx += stride) {
    size_t g = idx / per, off = idx - g * per;
    const float* src = (g == 0) ? Wf : (g == 1) ? Wi : (g == 2) ? Wc : Wo;
    Wout[idx] = (bf16_t)src[off];
  }
  for (size_t idx = i0; idx < 4 * D_LAT; idx += stride) {
    size_t g = idx >> 10, off = idx & 1023;
    const float* src = (g == 0) ? bf_ : (g == 1) ? bi_ : (g == 2) ? bc_ : bo_;
    bout[idx] = src[off];
  }
}

#if __has_builtin(__builtin_amdgcn_tensor_load_to_lds) && __has_builtin(__builtin_amdgcn_s_wait_tensorcnt)
#define LSTM455_TDM 1
#pragma message("lstm455: TDM tensor_load_to_lds path ENABLED")
#else
#define LSTM455_TDM 0
#pragma message("lstm455: TDM builtins missing -> scalar copy fallback")
#endif

// ---------------- persistent fused LSTM ----------------
// LDS layout (dynamic, 320 KB total):
//   [0      .. 131072) : Hs  = h_{t-1} staging, 64 x 1024 bf16
//   [131072 .. 327680) : Ws  = weight slice, 4 gates x 16 rows x 1536 bf16 (L2->LDS once)
__global__ void lstm455_persistent(const bf16_t* __restrict__ xbf,
                                   const bf16_t* __restrict__ Wbf,
                                   const float*  __restrict__ bias,
                                   bf16_t* __restrict__ Hbuf,   // [2][64][1024] bf16 (ping-pong)
                                   float*  __restrict__ out,    // [512][64][1024] f32
                                   unsigned* __restrict__ bar)  // [0]=count, [1]=generation
{
  extern __shared__ char smem[];
  bf16_t* Hs = (bf16_t*)smem;
  bf16_t* Ws = (bf16_t*)(smem + 131072);

  const int tid   = threadIdx.x;
  const int lane  = tid & 31;
  const int wid   = tid >> 5;        // wave id 0..3 -> batch block
  const int wg    = blockIdx.x;      // 0..63 -> latent slice
  const int mbase = wid * 16;
  const int lrow  = lane & 15;
  const int lhalf = lane >> 4;       // 0/1: selects contiguous K half [16*lhalf, +16) for A and B
  const int khx   = lhalf * 16;

  // ---- one-time: pull this WG's weight slice into LDS (192 KB, L2 read once) ----
  {
    const int chunksPerRow = (KTOT * 2) / 16;    // 192 x 16B per row
    for (int idx = tid; idx < 64 * chunksPerRow; idx += NTHREADS) {
      int r = idx / chunksPerRow;
      int ch = idx - r * chunksPerRow;
      int g = r >> 4, c = r & 15;
      const uint4* src = (const uint4*)(Wbf + (size_t)(g * D_LAT + wg * SLICE + c) * KTOT) + ch;
      uint4* dst = (uint4*)(Ws + (size_t)r * KTOT) + ch;
      *dst = *src;
    }
  }
  __syncthreads();

  // per-lane bias (N = lane%16 column of each gate tile), constant over t
  float bgate[4];
#pragma unroll
  for (int g = 0; g < 4; ++g) bgate[g] = bias[g * D_LAT + wg * SLICE + lrow];

  // per-wave B-row base pointers into the LDS weight slice (constant over t)
  const bf16_t* wb0 = Ws + (size_t)(0 * 16 + lrow) * KTOT + lhalf * 16;
  const bf16_t* wb1 = Ws + (size_t)(1 * 16 + lrow) * KTOT + lhalf * 16;
  const bf16_t* wb2 = Ws + (size_t)(2 * 16 + lrow) * KTOT + lhalf * 16;
  const bf16_t* wb3 = Ws + (size_t)(3 * 16 + lrow) * KTOT + lhalf * 16;

  v8f cstate = {};   // cell state lives in registers for all 512 steps

  for (int t = 0; t < T_STEPS; ++t) {
    const bf16_t* Hsrc = Hbuf + (size_t)(t & 1) * (BATCH * D_LAT);

    // ---- stage h_{t-1} (128 KB) global -> LDS via Tensor Data Mover ----
#if LSTM455_TDM
    if (wid == 0) {                       // wave-uniform branch: only wave 0 issues TDM
      uint64_t ga = (uint64_t)(uintptr_t)Hsrc;
      u32x4 g0;
      g0.x = 1u;                                            // count=1, user D#
      g0.y = 0u;                                            // lds_addr: Hs at base of dyn-LDS
      g0.z = (unsigned)(ga & 0xFFFFFFFFu);                  // global_addr[31:0]
      g0.w = (unsigned)((ga >> 32) & 0x01FFFFFFu) | 0x80000000u; // addr[56:32] | type=2
      i32x8 g1;
      g1[0] = (1 << 16);                                    // data_size=1 (2B), mask/flags=0
      g1[1] = (int)((D_LAT & 0xFFFF) << 16);                // tensor_dim0 lo16 @ [63:48]
      g1[2] = (int)((BATCH & 0xFFFF) << 16);                // dim0 hi=0, tensor_dim1 lo16
      g1[3] = (int)(D_LAT << 16);                           // dim1 hi=0, tile_dim0=1024
      g1[4] = BATCH;                                        // tile_dim1=64, tile_dim2=0
      g1[5] = D_LAT;                                        // tensor_dim0_stride lo32
      g1[6] = 0;                                            // stride0 hi, stride1 lo (2D: 0)
      g1[7] = 0;
      i32x4 gz = {};
# if __clang_major__ >= 23
      i32x8 gz8 = {};
      __builtin_amdgcn_tensor_load_to_lds(g0, g1, gz, gz, gz8, 0);
# else
      __builtin_amdgcn_tensor_load_to_lds(g0, g1, gz, gz, 0);
# endif
      __builtin_amdgcn_s_wait_tensorcnt(0);
    }
    __syncthreads();
#else
    for (int idx = tid; idx < (BATCH * D_LAT * 2) / 16; idx += NTHREADS)
      ((uint4*)Hs)[idx] = ((const uint4*)Hsrc)[idx];
    __syncthreads();
#endif

    v8f acc[4] = {{}, {}, {}, {}};

    // =========== software-pipelined K sweep, x part (K = 0..511) ===========
    const bf16_t* xrow = xbf + ((size_t)t * BATCH + mbase + lrow) * D_INP + khx;
    if (t + 1 < T_STEPS) __builtin_prefetch(xrow + (size_t)BATCH * D_INP, 0, 0);
    {
      v16bf a_cur = load_a16(xrow);
      v16bf b0 = load_b16(wb0), b1 = load_b16(wb1), b2 = load_b16(wb2), b3 = load_b16(wb3);
#pragma unroll 5
      for (int k0 = 0; k0 < D_INP - 32; k0 += 32) {
        // issue next-iteration loads before this iteration's matrix ops
        v16bf a_nxt = load_a16(xrow + k0 + 32);
        v16bf n0 = load_b16(wb0 + k0 + 32);
        v16bf n1 = load_b16(wb1 + k0 + 32);
        v16bf n2 = load_b16(wb2 + k0 + 32);
        v16bf n3 = load_b16(wb3 + k0 + 32);
        acc[0] = wmma_bf16<false>(a_cur, b0, acc[0]);
        acc[1] = wmma_bf16<true >(a_cur, b1, acc[1]);   // same A as previous wmma
        acc[2] = wmma_bf16<true >(a_cur, b2, acc[2]);
        acc[3] = wmma_bf16<true >(a_cur, b3, acc[3]);
        a_cur = a_nxt; b0 = n0; b1 = n1; b2 = n2; b3 = n3;
      }
      acc[0] = wmma_bf16<false>(a_cur, b0, acc[0]);
      acc[1] = wmma_bf16<true >(a_cur, b1, acc[1]);
      acc[2] = wmma_bf16<true >(a_cur, b2, acc[2]);
      acc[3] = wmma_bf16<true >(a_cur, b3, acc[3]);
    }

    // =========== software-pipelined K sweep, h part (K = 512..1535) ===========
    {
      const bf16_t* hrow = Hs + (size_t)(mbase + lrow) * D_LAT + khx;
      const bf16_t* hb0 = wb0 + D_INP;
      const bf16_t* hb1 = wb1 + D_INP;
      const bf16_t* hb2 = wb2 + D_INP;
      const bf16_t* hb3 = wb3 + D_INP;
      v16bf a_cur = load_a16(hrow);
      v16bf b0 = load_b16(hb0), b1 = load_b16(hb1), b2 = load_b16(hb2), b3 = load_b16(hb3);
#pragma unroll 4
      for (int k0 = 0; k0 < D_LAT - 32; k0 += 32) {
        v16bf a_nxt = load_a16(hrow + k0 + 32);
        v16bf n0 = load_b16(hb0 + k0 + 32);
        v16bf n1 = load_b16(hb1 + k0 + 32);
        v16bf n2 = load_b16(hb2 + k0 + 32);
        v16bf n3 = load_b16(hb3 + k0 + 32);
        acc[0] = wmma_bf16<false>(a_cur, b0, acc[0]);
        acc[1] = wmma_bf16<true >(a_cur, b1, acc[1]);
        acc[2] = wmma_bf16<true >(a_cur, b2, acc[2]);
        acc[3] = wmma_bf16<true >(a_cur, b3, acc[3]);
        a_cur = a_nxt; b0 = n0; b1 = n1; b2 = n2; b3 = n3;
      }
      acc[0] = wmma_bf16<false>(a_cur, b0, acc[0]);
      acc[1] = wmma_bf16<true >(a_cur, b1, acc[1]);
      acc[2] = wmma_bf16<true >(a_cur, b2, acc[2]);
      acc[3] = wmma_bf16<true >(a_cur, b3, acc[3]);
    }

    // ---- gates, cell update (c in registers), emit h ----
    bf16_t* Hdst = Hbuf + (size_t)((t + 1) & 1) * (BATCH * D_LAT);
    float* orow  = out + (size_t)t * (BATCH * D_LAT);
    const int lcol = wg * SLICE + lrow;
#pragma unroll
    for (int r = 0; r < 8; ++r) {
      float gf = fast_sigmoid(acc[0][r] + bgate[0]);
      float gi = fast_sigmoid(acc[1][r] + bgate[1]);
      float gc = fast_tanh   (acc[2][r] + bgate[2]);
      float go = fast_sigmoid(acc[3][r] + bgate[3]);
      float cn = gf * cstate[r] + gi * gc;
      cstate[r] = cn;
      float hn = go * fast_tanh(cn);
      int brow = mbase + r + lhalf * 8;          // C-tile row mapping per ISA
      Hdst[(size_t)brow * D_LAT + lcol] = (bf16_t)hn;
      orow[(size_t)brow * D_LAT + lcol] = hn;
    }

    // ---- device-wide step barrier (generation counter) ----
    __threadfence();
    __syncthreads();
    if (tid == 0) {
      unsigned prev = __hip_atomic_fetch_add(&bar[0], 1u, __ATOMIC_ACQ_REL, __HIP_MEMORY_SCOPE_AGENT);
      if (prev == (unsigned)(gridDim.x - 1)) {
        __hip_atomic_store(&bar[0], 0u, __ATOMIC_RELAXED, __HIP_MEMORY_SCOPE_AGENT);
        __hip_atomic_fetch_add(&bar[1], 1u, __ATOMIC_RELEASE, __HIP_MEMORY_SCOPE_AGENT);
      } else {
        while (__hip_atomic_load(&bar[1], __ATOMIC_ACQUIRE, __HIP_MEMORY_SCOPE_AGENT) <
               (unsigned)(t + 1)) {
          __builtin_amdgcn_s_sleep(2);
        }
      }
    }
    __syncthreads();
  }
}

// ---------------- launch ----------------
extern "C" void kernel_launch(void* const* d_in, const int* in_sizes, int n_in,
                              void* d_out, int out_size, void* d_ws, size_t ws_size,
                              hipStream_t stream) {
  (void)in_sizes; (void)n_in; (void)out_size; (void)ws_size;
  const float* x   = (const float*)d_in[0];
  const float* Wf  = (const float*)d_in[1];
  const float* bf_ = (const float*)d_in[2];
  const float* Wi  = (const float*)d_in[3];
  const float* bi_ = (const float*)d_in[4];
  const float* Wc  = (const float*)d_in[5];
  const float* bc_ = (const float*)d_in[6];
  const float* Wo  = (const float*)d_in[7];
  const float* bo_ = (const float*)d_in[8];
  float* out = (float*)d_out;

  char* ws = (char*)d_ws;
  size_t off = 0;
  unsigned* bar  = (unsigned*)(ws + off); off += 256;
  float*    bias = (float*)(ws + off);    off += 4096 * sizeof(float);
  off = (off + 255) & ~(size_t)255;
  bf16_t* Hbuf = (bf16_t*)(ws + off);     off += (size_t)2 * BATCH * D_LAT * sizeof(bf16_t);
  off = (off + 255) & ~(size_t)255;
  bf16_t* Wbf  = (bf16_t*)(ws + off);     off += (size_t)4096 * KTOT * sizeof(bf16_t);
  off = (off + 255) & ~(size_t)255;
  bf16_t* xbf  = (bf16_t*)(ws + off);     off += (size_t)T_STEPS * BATCH * D_INP * sizeof(bf16_t);

  hipMemsetAsync(bar, 0, 256, stream);
  hipMemsetAsync(Hbuf, 0, (size_t)2 * BATCH * D_LAT * sizeof(bf16_t), stream);

  lstm455_cvt_x<<<4096, 256, 0, stream>>>(x, xbf, (T_STEPS * BATCH * D_INP) / 4);
  lstm455_pack_w<<<4096, 256, 0, stream>>>(Wf, Wi, Wc, Wo, bf_, bi_, bc_, bo_, Wbf, bias);

  size_t shmem = 131072 + (size_t)64 * KTOT * sizeof(bf16_t);  // 128KB H + 192KB W = 320KB
  lstm455_persistent<<<NWG, NTHREADS, shmem, stream>>>(xbf, Wbf, bias, Hbuf, out, bar);
}
